// MulLayer_76330158784725
// MI455X (gfx1250) — matrix operand order repaired and checked by reading, not compiled, not run on hardware
//
#include <hip/hip_runtime.h>

typedef __attribute__((ext_vector_type(16))) _Float16 v16h;
typedef __attribute__((ext_vector_type(8)))  _Float16 v8h;
typedef __attribute__((ext_vector_type(8)))  float    v8f;

#define KREG 9
#define HH 192
#define WW 192
#define NPIX (HH*WW)
#define CCH 256

// ---------------- mask preprocessing ----------------
__global__ void k_maskbits(const int* __restrict__ mask, int* __restrict__ mbits) {
  int n = blockIdx.x * 256 + threadIdx.x;
  if (n >= NPIX) return;
  int b = 0;
#pragma unroll
  for (int k = 0; k < KREG; ++k) b |= (mask[(size_t)k * NPIX + n] > 0) ? (1 << k) : 0;
  mbits[n] = b;
}

__global__ void k_counts(const int* __restrict__ mbits, float* __restrict__ cnt) {
  __shared__ float sh[256];
  int k = blockIdx.x;
  float s = 0.f;
  for (int n = threadIdx.x; n < NPIX; n += 256) s += (float)((mbits[n] >> k) & 1);
  sh[threadIdx.x] = s;
  __syncthreads();
  for (int o = 128; o > 0; o >>= 1) {
    if (threadIdx.x < o) sh[threadIdx.x] += sh[threadIdx.x + o];
    __syncthreads();
  }
  if (threadIdx.x == 0) cnt[k] = sh[0];
}

// sums[k*C + c] = sum_n mask[k,n] * x[c,n]   (one block per channel)
__global__ void k_chansums(const float* __restrict__ x, const int* __restrict__ mbits,
                           float* __restrict__ sums) {
  int c = blockIdx.x;
  float acc[KREG];
#pragma unroll
  for (int k = 0; k < KREG; ++k) acc[k] = 0.f;
  for (int n = threadIdx.x; n < NPIX; n += 256) {
    float v = x[(size_t)c * NPIX + n];
    int b = mbits[n];
#pragma unroll
    for (int k = 0; k < KREG; ++k) if ((b >> k) & 1) acc[k] += v;
  }
  __shared__ float sh[KREG];
  if (threadIdx.x < KREG) sh[threadIdx.x] = 0.f;
  __syncthreads();
#pragma unroll
  for (int k = 0; k < KREG; ++k) atomicAdd(&sh[k], acc[k]);
  __syncthreads();
  if (threadIdx.x < KREG) sums[(size_t)threadIdx.x * CCH + c] = sh[threadIdx.x];
}

// last region k containing pixel n with cntA[k]>=10 && cntB[k]>=10, else -1
__global__ void k_rid(const int* __restrict__ mbits, const float* __restrict__ cntA,
                      const float* __restrict__ cntB, int* __restrict__ rid) {
  int n = blockIdx.x * 256 + threadIdx.x;
  if (n >= NPIX) return;
  int b = mbits[n], r = -1;
#pragma unroll
  for (int k = 0; k < KREG; ++k)
    if (((b >> k) & 1) && cntA[k] >= 10.f && cntB[k] >= 10.f) r = k;
  rid[n] = r;
}

__global__ void k_zero(float* __restrict__ p, int n) {
  int i = blockIdx.x * 256 + threadIdx.x;
  if (i < n) p[i] = 0.f;
}

// ---------------- layout transforms ----------------
// [R, NPIX] f32 -> [NPIX, R] f16, optional per-region mean subtraction
__global__ void k_transpose_cast(const float* __restrict__ in, _Float16* __restrict__ out,
                                 int R, const float* __restrict__ sums,
                                 const float* __restrict__ cnt, const int* __restrict__ rid) {
  __shared__ float tile[32][33];
  int n0 = blockIdx.x * 32, c0 = blockIdx.y * 32;
  for (int i = threadIdx.y; i < 32; i += 8)
    tile[i][threadIdx.x] = in[(size_t)(c0 + i) * NPIX + n0 + threadIdx.x];
  __syncthreads();
  for (int i = threadIdx.y; i < 32; i += 8) {
    int n = n0 + i, c = c0 + threadIdx.x;
    float v = tile[threadIdx.x][i];
    if (sums) {
      int r = rid[n];
      if (r >= 0) v -= sums[(size_t)r * R + c] / fmaxf(cnt[r], 1.f);
    }
    out[(size_t)n * R + c] = (_Float16)v;
  }
}

// weights [Co, Ci, kt] f32 -> wq[t][m][c] f16 (tap-major, K-contiguous)
__global__ void k_wconvert(const float* __restrict__ w, _Float16* __restrict__ wq,
                           int Co, int Ci, int kt) {
  int idx = blockIdx.x * 256 + threadIdx.x;
  int tot = Co * Ci * kt;
  if (idx >= tot) return;
  int t = idx / (Co * Ci), rem = idx % (Co * Ci);
  int m = rem / Ci, c = rem % Ci;
  wq[idx] = (_Float16)w[((size_t)m * Ci + c) * kt + t];
}

// ---------------- WMMA implicit-GEMM conv (3x3 or 1x1) ----------------
__device__ __forceinline__ v16h ldA(const _Float16* p) {
  // A-fragment 16x32 f16: lane m=lane&15; elems 0..15 = K(c0+16*hi+e)
  v8h lo = *(const v8h*)p;
  v8h hh = *(const v8h*)(p + 8);
  v16h r;
#pragma unroll
  for (int i = 0; i < 8; ++i) { r[i] = lo[i]; r[i + 8] = hh[i]; }
  return r;
}

__device__ __forceinline__ v8f wmma16(v16h a, v16h b, v8f c) {
  return __builtin_amdgcn_wmma_f32_16x16x32_f16(false, a, false, b, (short)0, c, false, false);
}

// in_t: [NPIX, CIN] f16 channels-last; wq: [tap][COUT][CIN] f16; out: [COUT, NPIX] f32
// wave tile: 32 (M) x 32 (N) = 4 accumulators; 4 waves per block.
template <int CIN, int COUT, int KS, int RELU>
__global__ __launch_bounds__(128)
void k_conv_wmma(const _Float16* __restrict__ in_t, const _Float16* __restrict__ wq,
                 const float* __restrict__ bias, float* __restrict__ out) {
  constexpr int PAD = KS / 2;
  const int lane = threadIdx.x & 31;
  const int wave = threadIdx.x >> 5;
  const int n0 = (blockIdx.x * 4 + wave) * 32;   // 32 | WW: never crosses a row
  const int m0 = blockIdx.y * 32;
  const int y  = n0 / WW;                        // compile-time divisor
  const int x0 = n0 % WW;
  const int col = lane & 15;
  const int hi  = lane >> 4;

  v8f acc00 = {}, acc01 = {}, acc10 = {}, acc11 = {};
  const _Float16* a0 = wq + (size_t)(m0 + col) * CIN + 16 * hi;
  const _Float16* a1 = a0 + (size_t)16 * CIN;
#pragma unroll 1
  for (int t = 0; t < KS * KS; ++t) {
    const int dy = t / KS - PAD, dx = t % KS - PAD;
    const int yy = y + dy;
    const int xa = x0 + col + dx;
    const int xb = xa + 16;
    const bool rok = (yy >= 0) && (yy < HH);
    const bool ok0 = rok && (xa >= 0) && (xa < WW);
    const bool ok1 = rok && (xb >= 0) && (xb < WW);
    const _Float16* b0 = in_t + (size_t)(yy * WW + xa) * CIN + 16 * hi;
    const _Float16* b1 = in_t + (size_t)(yy * WW + xb) * CIN + 16 * hi;
    const _Float16* ap0 = a0 + (size_t)t * COUT * CIN;
    const _Float16* ap1 = a1 + (size_t)t * COUT * CIN;
#pragma unroll
    for (int c0 = 0; c0 < CIN; c0 += 32) {
      v16h av0 = ldA(ap0 + c0);
      v16h av1 = ldA(ap1 + c0);
      v16h bv0 = {}, bv1 = {};
      if (ok0) bv0 = *(const v16h*)(b0 + c0);    // K = c0 + 16*hi + [0..15]
      if (ok1) bv1 = *(const v16h*)(b1 + c0);
      acc00 = wmma16(av0, bv0, acc00);
      acc01 = wmma16(av0, bv1, acc01);
      acc10 = wmma16(av1, bv0, acc10);
      acc11 = wmma16(av1, bv1, acc11);
    }
  }
  const int na = n0 + col, nb = na + 16;
#pragma unroll
  for (int rr = 0; rr < 8; ++rr) {
    int mlo = m0 + rr + 8 * hi;
    int mhi = mlo + 16;
    float blo = bias[mlo], bhi = bias[mhi];
    float v00 = acc00[rr] + blo, v01 = acc01[rr] + blo;
    float v10 = acc10[rr] + bhi, v11 = acc11[rr] + bhi;
    if (RELU) {
      v00 = fmaxf(v00, 0.f); v01 = fmaxf(v01, 0.f);
      v10 = fmaxf(v10, 0.f); v11 = fmaxf(v11, 0.f);
    }
    out[(size_t)mlo * NPIX + na] = v00;
    out[(size_t)mlo * NPIX + nb] = v01;
    out[(size_t)mhi * NPIX + na] = v10;
    out[(size_t)mhi * NPIX + nb] = v11;
  }
}

// ---------------- masked covariance ----------------
#define COVCHUNK 256
__global__ __launch_bounds__(256)
void k_cov(const float* __restrict__ feat, const int* __restrict__ mbits,
           float* __restrict__ cov) {
  __shared__ float sf[32][COVCHUNK + 1];
  int k = blockIdx.y;
  int n = blockIdx.x * COVCHUNK + threadIdx.x;
  float m = ((mbits[n] >> k) & 1) ? 1.f : 0.f;
#pragma unroll
  for (int a = 0; a < 32; ++a)
    sf[a][threadIdx.x] = feat[(size_t)a * NPIX + n] * m;
  __syncthreads();
#pragma unroll
  for (int j = 0; j < 4; ++j) {
    int idx = threadIdx.x + 256 * j;
    int a = idx >> 5, b = idx & 31;
    float s = 0.f;
    for (int i = 0; i < COVCHUNK; ++i) s += sf[a][i] * sf[b][i];
    atomicAdd(&cov[k * 1024 + idx], s);
  }
}

// trans[k,j] = sum_i (cov[k,i]/max(cnt,1)) * fcw[j,i] + fcb[j]
__global__ void k_fc(const float* __restrict__ cov, const float* __restrict__ cnt,
                     const float* __restrict__ fcw, const float* __restrict__ fcb,
                     float* __restrict__ outMat) {
  int idx = blockIdx.x * 256 + threadIdx.x;
  if (idx >= KREG * 1024) return;
  int k = idx >> 10, j = idx & 1023;
  float inv = 1.f / fmaxf(cnt[k], 1.f);
  const float* cr = cov + k * 1024;
  const float* wr = fcw + (size_t)j * 1024;
  float s = 0.f;
  for (int i = 0; i < 1024; ++i) s += cr[i] * inv * wr[i];
  outMat[idx] = s + fcb[j];
}

// T[k] = sMat[k] (32x32) @ cMat[k] (32x32)
__global__ __launch_bounds__(1024)
void k_matT(const float* __restrict__ sM, const float* __restrict__ cM, float* __restrict__ T) {
  __shared__ float As[1024], Bs[1024];
  int k = blockIdx.x, t = threadIdx.x;
  As[t] = sM[k * 1024 + t];
  Bs[t] = cM[k * 1024 + t];
  __syncthreads();
  int a = t >> 5, b = t & 31;
  float s = 0.f;
#pragma unroll
  for (int c = 0; c < 32; ++c) s += As[a * 32 + c] * Bs[c * 32 + b];
  T[k * 1024 + t] = s;
}

// tf_t[n*32+a] = (rid<0) ? ccf[a,n] : sum_b T[rid][a,b]*ccf[b,n]   (f16, channels-last)
__global__ void k_transfeat(const float* __restrict__ ccf, const float* __restrict__ T,
                            const int* __restrict__ rid, _Float16* __restrict__ tf_t) {
  __shared__ float sc[32][9];
  int n0 = blockIdx.x * 8;
  int ch = threadIdx.x >> 3, p = threadIdx.x & 7;
  sc[ch][p] = ccf[(size_t)ch * NPIX + n0 + p];
  __syncthreads();
  int a = threadIdx.x & 31, p2 = threadIdx.x >> 5;
  int n = n0 + p2;
  int r = rid[n];
  float v;
  if (r < 0) v = sc[a][p2];
  else {
    v = 0.f;
    const float* Tr = T + r * 1024 + a * 32;
#pragma unroll
    for (int b = 0; b < 32; ++b) v += Tr[b] * sc[b][p2];
  }
  tf_t[(size_t)n * 32 + a] = (_Float16)v;
}

// out[c,n] += sMeans[rid[n]][c] if in valid region
__global__ void k_finaladd(float* __restrict__ out, const int* __restrict__ rid,
                           const float* __restrict__ sums_s, const float* __restrict__ cnt_s) {
  size_t idx = (size_t)blockIdx.x * 256 + threadIdx.x;
  if (idx >= (size_t)CCH * NPIX) return;
  int c = (int)(idx / NPIX), n = (int)(idx % NPIX);
  int r = rid[n];
  if (r >= 0) out[idx] += sums_s[(size_t)r * CCH + c] / fmaxf(cnt_s[r], 1.f);
}

// ---------------- host launcher ----------------
extern "C" void kernel_launch(void* const* d_in, const int* in_sizes, int n_in,
                              void* d_out, int out_size, void* d_ws, size_t ws_size,
                              hipStream_t stream) {
  (void)in_sizes; (void)n_in; (void)out_size; (void)ws_size;
  const int N = NPIX;
  const float* cF = (const float*)d_in[0];
  const float* sF = (const float*)d_in[1];
  const int* cmasks = (const int*)d_in[2];
  const int* smasks = (const int*)d_in[3];

  char* base = (char*)d_ws;
  size_t off = 0;
  auto alloc = [&](size_t bytes) -> void* {
    void* p = base + off;
    off = (off + bytes + 255) & ~(size_t)255;
    return p;
  };
  float* cnt[2]  = { (float*)alloc(KREG * 4), (float*)alloc(KREG * 4) };
  int*   mbits[2] = { (int*)alloc((size_t)N * 4), (int*)alloc((size_t)N * 4) };
  float* sums[2] = { (float*)alloc((size_t)KREG * CCH * 4), (float*)alloc((size_t)KREG * CCH * 4) };
  int*   rid[2]  = { (int*)alloc((size_t)N * 4), (int*)alloc((size_t)N * 4) };
  int*   rid_cc  = (int*)alloc((size_t)N * 4);
  float* Mat[2]  = { (float*)alloc(KREG * 1024 * 4), (float*)alloc(KREG * 1024 * 4) };
  float* Tm      = (float*)alloc(KREG * 1024 * 4);
  float* cov     = (float*)alloc(KREG * 1024 * 4);
  _Float16* w1h  = (_Float16*)alloc((size_t)128 * 256 * 9 * 2);
  _Float16* w2h  = (_Float16*)alloc((size_t)64 * 128 * 9 * 2);
  _Float16* w3h  = (_Float16*)alloc((size_t)32 * 64 * 9 * 2);
  _Float16* cmph = (_Float16*)alloc((size_t)32 * 256 * 2);
  _Float16* uzh  = (_Float16*)alloc((size_t)256 * 32 * 2);
  _Float16* fsm_t = (_Float16*)alloc((size_t)N * 256 * 2);
  float*    h1    = (float*)alloc((size_t)128 * N * 4);
  _Float16* h1_t  = (_Float16*)alloc((size_t)N * 128 * 2);
  float*    h2    = (float*)alloc((size_t)64 * N * 4);
  _Float16* h2_t  = (_Float16*)alloc((size_t)N * 64 * 2);
  float*    feat  = (float*)alloc((size_t)32 * N * 4);
  float*    ccf   = h2;                 // alias: h2 dead after h2_t built
  _Float16* tf_t  = (_Float16*)h1;      // alias: h1 dead after h1_t built
  float*    out   = (float*)d_out;

  k_maskbits<<<N / 256, 256, 0, stream>>>(smasks, mbits[0]);
  k_maskbits<<<N / 256, 256, 0, stream>>>(cmasks, mbits[1]);

  dim3 tb(32, 8);
  const dim3 cgrid(N / 128, 0, 1);   // y set per-conv below via COUT/32
  for (int ph = 0; ph < 2; ++ph) {   // 0 = style (sF), 1 = content (cF)
    const float* x   = ph ? cF : sF;
    const float* c1w = (const float*)d_in[ph ? 12 : 4];
    const float* c1b = (const float*)d_in[ph ? 13 : 5];
    const float* c2w = (const float*)d_in[ph ? 14 : 6];
    const float* c2b = (const float*)d_in[ph ? 15 : 7];
    const float* c3w = (const float*)d_in[ph ? 16 : 8];
    const float* c3b = (const float*)d_in[ph ? 17 : 9];
    const float* fcw = (const float*)d_in[ph ? 18 : 10];
    const float* fcb = (const float*)d_in[ph ? 19 : 11];

    k_counts<<<KREG, 256, 0, stream>>>(mbits[ph], cnt[ph]);
    k_chansums<<<CCH, 256, 0, stream>>>(x, mbits[ph], sums[ph]);
    k_rid<<<N / 256, 256, 0, stream>>>(mbits[ph], cnt[ph], cnt[ph], rid[ph]);
    k_transpose_cast<<<dim3(N / 32, CCH / 32), tb, 0, stream>>>(x, fsm_t, CCH,
                                                                sums[ph], cnt[ph], rid[ph]);
    k_wconvert<<<128 * 256 * 9 / 256, 256, 0, stream>>>(c1w, w1h, 128, 256, 9);
    k_wconvert<<<64 * 128 * 9 / 256, 256, 0, stream>>>(c2w, w2h, 64, 128, 9);
    k_wconvert<<<32 * 64 * 9 / 256, 256, 0, stream>>>(c3w, w3h, 32, 64, 9);

    k_conv_wmma<256, 128, 3, 1><<<dim3(N / 128, 128 / 32), 128, 0, stream>>>(fsm_t, w1h, c1b, h1);
    k_transpose_cast<<<dim3(N / 32, 128 / 32), tb, 0, stream>>>(h1, h1_t, 128,
                                                                nullptr, nullptr, nullptr);
    k_conv_wmma<128, 64, 3, 1><<<dim3(N / 128, 64 / 32), 128, 0, stream>>>(h1_t, w2h, c2b, h2);
    k_transpose_cast<<<dim3(N / 32, 64 / 32), tb, 0, stream>>>(h2, h2_t, 64,
                                                               nullptr, nullptr, nullptr);
    k_conv_wmma<64, 32, 3, 0><<<dim3(N / 128, 32 / 32), 128, 0, stream>>>(h2_t, w3h, c3b, feat);
    k_zero<<<KREG * 1024 / 256, 256, 0, stream>>>(cov, KREG * 1024);
    k_cov<<<dim3(N / COVCHUNK, KREG), 256, 0, stream>>>(feat, mbits[ph], cov);
    k_fc<<<KREG * 1024 / 256, 256, 0, stream>>>(cov, cnt[ph], fcw, fcb, Mat[ph]);
  }

  // composition: validity = svalid & cvalid over cmask regions
  k_rid<<<N / 256, 256, 0, stream>>>(mbits[1], cnt[0], cnt[1], rid_cc);
  k_wconvert<<<32 * 256 / 256, 256, 0, stream>>>((const float*)d_in[20], cmph, 32, 256, 1);
  k_conv_wmma<256, 32, 1, 0><<<dim3(N / 128, 32 / 32), 128, 0, stream>>>(
      fsm_t, cmph, (const float*)d_in[21], ccf);
  k_matT<<<KREG, 1024, 0, stream>>>(Mat[0], Mat[1], Tm);
  k_transfeat<<<N / 8, 256, 0, stream>>>(ccf, Tm, rid_cc, tf_t);
  k_wconvert<<<256 * 32 / 256, 256, 0, stream>>>((const float*)d_in[22], uzh, 256, 32, 1);
  k_conv_wmma<32, 256, 1, 0><<<dim3(N / 128, 256 / 32), 128, 0, stream>>>(
      tf_t, uzh, (const float*)d_in[23], out);
  k_finaladd<<<(int)(((size_t)CCH * N) / 256), 256, 0, stream>>>(out, rid_cc, sums[0], cnt[0]);
}
